// Nerf_29403346108731
// MI455X (gfx1250) — matrix operand order repaired and checked by reading, not compiled, not run on hardware
//
#include <hip/hip_runtime.h>
#include <cstdint>

#define THREADS 256
#define RAYS_WG 128
#define HS   264     // H row stride (f16 elems), 528B -> b128 conflict-free
#define XS   72      // X row stride (63 -> pad 64)
#define DSTR 40      // D row stride (27 -> pad 32)
#define WSTR 264     // weight row stride (K=256 layers)

#define OFF_H  0
#define OFF_X  (RAYS_WG * HS * 2)            // 67584
#define OFF_D  (OFF_X + RAYS_WG * XS * 2)    // 86016
#define OFF_W  (OFF_D + RAYS_WG * DSTR * 2)  // 96256
#define OFF_WX (OFF_W + 272 * WSTR * 2)      // 239872
#define LDS_BYTES (OFF_WX + 256 * XS * 2)    // 276736  (< 320KB/WGP)

// ---- f16 weight-image layout in d_ws (element offsets) ----
#define IMG_W1_0   0
#define IMG_W1_1   (IMG_W1_0  + 256 * 72)
#define IMG_W1_2   (IMG_W1_1  + 256 * 264)
#define IMG_W1_3   (IMG_W1_2  + 256 * 264)
#define IMG_W1_4   (IMG_W1_3  + 256 * 264)
#define IMG_W2_0M  (IMG_W1_4  + 256 * 264)
#define IMG_W2_0X  (IMG_W2_0M + 256 * 264)
#define IMG_W2_1   (IMG_W2_0X + 256 * 72)
#define IMG_W2_2   (IMG_W2_1  + 256 * 264)
#define IMG_W2_3   (IMG_W2_2  + 256 * 264)
#define IMG_WR0M   (IMG_W2_3  + 272 * 264)
#define IMG_WR0X   (IMG_WR0M  + 128 * 264)
#define IMG_TOTAL  (IMG_WR0X  + 128 * 40)
#define WS_NEEDED  ((size_t)IMG_TOTAL * 2)

typedef __attribute__((ext_vector_type(16))) _Float16 v16h;
typedef __attribute__((ext_vector_type(8)))  _Float16 v8h;
typedef __attribute__((ext_vector_type(4)))  _Float16 v4h;
typedef __attribute__((ext_vector_type(8)))  float    v8f;
typedef __attribute__((ext_vector_type(4)))  int      v4i;

// AS-qualified int4 for the async-to-LDS builtin (AS1 = global, AS3 = LDS)
typedef __attribute__((address_space(1))) v4i GV4;
typedef __attribute__((address_space(3))) v4i LV4;

struct Params {
  const float *xyz, *dir;
  const float *w1[5], *b1[5];
  const float *w2[4], *b2[4];
  const float *wr0, *br0, *wr1, *br1;
  const _Float16 *wimg;   // pre-converted f16 weight images (may be null)
  float *out_c, *out_sigma;
};

// ---------------- async global->LDS staging (CDNA5 path) ----------------
#if defined(__has_builtin)
#if __has_builtin(__builtin_amdgcn_global_load_async_to_lds_b128)
#define HAVE_ASYNC_LDS 1
#endif
#endif

#if defined(__has_builtin)
#if __has_builtin(__builtin_amdgcn_s_wait_asynccnt)
#define WAIT_ASYNC() __builtin_amdgcn_s_wait_asynccnt(0)
#else
#define WAIT_ASYNC() asm volatile("s_wait_asynccnt 0x0" ::: "memory")
#endif
#else
#define WAIT_ASYNC() asm volatile("s_wait_asynccnt 0x0" ::: "memory")
#endif

__device__ __forceinline__ void copy16(_Float16* lds, const _Float16* g) {
#ifdef HAVE_ASYNC_LDS
  __builtin_amdgcn_global_load_async_to_lds_b128((GV4*)g, (LV4*)lds, 0, 0);
#else
  *(v8h*)lds = *(const v8h*)g;   // global_load_b128 + ds_store_b128
#endif
}

// Linear image copy: elems must be a multiple of 8 f16 (16 bytes).
__device__ __forceinline__ void stage_img(_Float16* lds, const _Float16* g,
                                          int elems, int tid) {
  const int n16 = elems >> 3;
  for (int i = tid; i < n16; i += THREADS) {
    copy16(lds + (i << 3), g + (i << 3));
  }
}

// ---------------- WMMA fragments ----------------
__device__ __forceinline__ v8f wmma_f16(v16h a, v16h b, v8f c) {
  return __builtin_amdgcn_wmma_f32_16x16x32_f16(
      false, a, false, b, (short)0, c, false, false);
}

// A: 16x32 f16 tile (rows = rays). lanes 0-15: K0..7 (v0-3) & K16..23 (v4-7);
// lanes 16-31: K8..15 & K24..31.
__device__ __forceinline__ v16h load_a(const _Float16* buf, int stride,
                                       int row0, int k0, int lane) {
  const int m = lane & 15, khi = lane >> 4;
  const _Float16* q = buf + (row0 + m) * stride + k0 + khi * 8;
  const v8h lo = *(const v8h*)q;
  const v8h hi = *(const v8h*)(q + 16);
  v16h r;
#pragma unroll
  for (int i = 0; i < 8; ++i) { r[i] = lo[i]; r[i + 8] = hi[i]; }
  return r;
}

// B: 32x16 (KxN). W row-major (out x in) so B[k][n] = W[n0+n][k].
__device__ __forceinline__ v16h load_b(const _Float16* wlds, int stride,
                                       int n0, int k0, int lane) {
  const int n = lane & 15, kb = (lane >> 4) << 3;
  const _Float16* q = wlds + (n0 + n) * stride + k0 + kb;
  const v8h lo = *(const v8h*)q;
  const v8h hi = *(const v8h*)(q + 16);
  v16h r;
#pragma unroll
  for (int i = 0; i < 8; ++i) { r[i] = lo[i]; r[i + 8] = hi[i]; }
  return r;
}

template <int NT>
__device__ __forceinline__ void zero_acc(v8f* acc) {
#pragma unroll
  for (int i = 0; i < NT; ++i)
    acc[i] = (v8f){0.f, 0.f, 0.f, 0.f, 0.f, 0.f, 0.f, 0.f};
}

template <int NT>
__device__ __forceinline__ void accumulate(v8f* acc, const _Float16* inbuf, int inStride,
                                           int row0, int K, const _Float16* wlds,
                                           int wStride, int lane) {
  for (int k0 = 0; k0 < K; k0 += 32) {
    const v16h a = load_a(inbuf, inStride, row0, k0, lane);
#pragma unroll
    for (int nt = 0; nt < NT; ++nt) {
      const v16h b = load_b(wlds, wStride, nt * 16, k0, lane);
      acc[nt] = wmma_f16(a, b, acc[nt]);
    }
  }
}

// C/D: lane holds col n=lane&15; VGPR r -> row r + 8*(lane>=16).
template <int NT>
__device__ __forceinline__ void writeback(const v8f* acc, _Float16* outb, int oStride,
                                          int row0, const float* bias, int lane,
                                          bool do_relu) {
  const int n = lane & 15, mb = (lane >> 4) << 3;
#pragma unroll
  for (int nt = 0; nt < NT; ++nt) {
    const float bv = bias[nt * 16 + n];
#pragma unroll
    for (int r = 0; r < 8; ++r) {
      float v = acc[nt][r] + bv;
      if (do_relu) v = fmaxf(v, 0.f);
      outb[(row0 + mb + r) * oStride + nt * 16 + n] = (_Float16)v;
    }
  }
}

// ---------------- fallback f32 staging (used only if ws too small) ------
__device__ __forceinline__ void stage_w(_Float16* dst, int dstStride, const float* src,
                                        int srcStride, int srcOff, int srcRows, int rows,
                                        int K, int kpShift, int tid) {
  const int Kp = 1 << kpShift;
  const int total = rows << kpShift;
  for (int idx = tid; idx < total; idx += THREADS) {
    const int r = idx >> kpShift;
    const int k = idx & (Kp - 1);
    const float v = (r < srcRows && k < K) ? src[r * srcStride + srcOff + k] : 0.f;
    dst[r * dstStride + k] = (_Float16)v;
  }
}

__device__ __forceinline__ void stage_w4(_Float16* dst, int dstStride, const float* src,
                                         int rows, int tid) {
  const int total = rows << 6;
  for (int idx = tid; idx < total; idx += THREADS) {
    const int r = idx >> 6, k4 = (idx & 63) << 2;
    const float4 v = *(const float4*)(src + (r << 8) + k4);
    v4h hv;
    hv[0] = (_Float16)v.x; hv[1] = (_Float16)v.y;
    hv[2] = (_Float16)v.z; hv[3] = (_Float16)v.w;
    *(v4h*)(dst + r * dstStride + k4) = hv;
  }
}

// ---------------- shared device body pieces ----------------
__device__ __forceinline__ void posenc_body(const Params& p, _Float16* Xb, _Float16* Db,
                                            int ray0, int tid) {
  const int r = tid >> 1, half = tid & 1;
  const int g = ray0 + r;
  const float px = p.xyz[g * 3 + 0], py = p.xyz[g * 3 + 1], pz = p.xyz[g * 3 + 2];
  _Float16* xr = Xb + r * XS;
  if (!half) {
    xr[0] = (_Float16)px; xr[1] = (_Float16)py; xr[2] = (_Float16)pz;
    xr[63] = (_Float16)0.f;
  }
#pragma unroll
  for (int i = 0; i < 5; ++i) {
    const int l = half * 5 + i;
    const float f = (float)(1u << l);
    float sx, cx, sy, cy, sz, cz;
    __sincosf(px * f, &sx, &cx);
    __sincosf(py * f, &sy, &cy);
    __sincosf(pz * f, &sz, &cz);
    _Float16* o = xr + 3 + 6 * l;
    o[0] = (_Float16)sx; o[1] = (_Float16)sy; o[2] = (_Float16)sz;
    o[3] = (_Float16)cx; o[4] = (_Float16)cy; o[5] = (_Float16)cz;
  }
  const float dx = p.dir[g * 3 + 0], dy = p.dir[g * 3 + 1], dz = p.dir[g * 3 + 2];
  _Float16* dr = Db + r * DSTR;
  if (!half) {
    dr[0] = (_Float16)dx; dr[1] = (_Float16)dy; dr[2] = (_Float16)dz;
#pragma unroll
    for (int k2 = 27; k2 < 32; ++k2) dr[k2] = (_Float16)0.f;
  }
#pragma unroll
  for (int i = 0; i < 2; ++i) {
    const int l = half * 2 + i;
    const float f = (float)(1u << l);
    float sx, cx, sy, cy, sz, cz;
    __sincosf(dx * f, &sx, &cx);
    __sincosf(dy * f, &sy, &cy);
    __sincosf(dz * f, &sz, &cz);
    _Float16* o = dr + 3 + 6 * l;
    o[0] = (_Float16)sx; o[1] = (_Float16)sy; o[2] = (_Float16)sz;
    o[3] = (_Float16)cx; o[4] = (_Float16)cy; o[5] = (_Float16)cz;
  }
}

__device__ __forceinline__ void sigma_and_head(const Params& p, _Float16* Hb, _Float16* Db,
                                               _Float16* Wb, _Float16* WXb,
                                               int ray0, int row0, int lane, int tid,
                                               bool f16w) {
  // layer 2_3 weights are already in Wb (incl. zero rows 257..271)
  {
    v8f acc[17]; zero_acc<17>(acc);
    accumulate<17>(acc, Hb, HS, row0, 256, Wb, WSTR, lane);
    writeback<16>(acc, Hb, HS, row0, p.b2[3], lane, false);
    if ((lane & 15) == 0) {
      const int mb = (lane >> 4) << 3;
      const float bv = p.b2[3][256];
#pragma unroll
      for (int r = 0; r < 8; ++r) {
        const float v = acc[16][r] + bv;
        p.out_sigma[ray0 + row0 + mb + r] = fmaxf(v, 0.f);
      }
    }
  }

  // radiance: [h(256) | d_emb(32)] -> 128, ReLU
  __syncthreads();
  if (f16w) {
    stage_img(Wb, p.wimg + IMG_WR0M, 128 * 264, tid);
    stage_img(WXb, p.wimg + IMG_WR0X, 128 * 40, tid);
#ifdef HAVE_ASYNC_LDS
    WAIT_ASYNC();
#endif
  } else {
    stage_w(Wb, WSTR, p.wr0, 283, 0, 128, 128, 256, 8, tid);
    stage_w(WXb, DSTR, p.wr0, 283, 256, 128, 128, 27, 5, tid);
  }
  __syncthreads();
  {
    v8f acc[8]; zero_acc<8>(acc);
    accumulate<8>(acc, Hb, HS, row0, 256, Wb, WSTR, lane);
    accumulate<8>(acc, Db, DSTR, row0, 32, WXb, DSTR, lane);
    writeback<8>(acc, Hb, HS, row0, p.br0, lane, true);
  }

  // head: 128 -> 3, sigmoid
  __syncthreads();
  for (int task = tid; task < RAYS_WG * 3; task += THREADS) {
    const int rl = task & (RAYS_WG - 1);
    const int ch = task >> 7;
    const _Float16* gp = Hb + rl * HS;
    const float* wp = p.wr1 + ch * 128;
    float s = 0.f;
#pragma unroll 4
    for (int k = 0; k < 128; ++k) s += (float)gp[k] * wp[k];
    s += p.br1[ch];
    p.out_c[(ray0 + rl) * 3 + ch] = 1.f / (1.f + __expf(-s));
  }
}

// ================= main kernel: pre-converted f16 weights =================
__global__ void __launch_bounds__(THREADS, 1)
nerf_fused_f16w(Params p) {
  extern __shared__ char smem[];
  _Float16* Hb  = (_Float16*)(smem + OFF_H);
  _Float16* Xb  = (_Float16*)(smem + OFF_X);
  _Float16* Db  = (_Float16*)(smem + OFF_D);
  _Float16* Wb  = (_Float16*)(smem + OFF_W);
  _Float16* WXb = (_Float16*)(smem + OFF_WX);

  const int tid  = threadIdx.x;
  const int lane = tid & 31;
  const int wave = tid >> 5;
  const int row0 = wave << 4;
  const int ray0 = blockIdx.x * RAYS_WG;

  posenc_body(p, Xb, Db, ray0, tid);

  // layer 1_0 : 63(pad64) -> 256 (image stride 72)
  __syncthreads();
  stage_img(Wb, p.wimg + IMG_W1_0, 256 * 72, tid);
#ifdef HAVE_ASYNC_LDS
  WAIT_ASYNC();
#endif
  __syncthreads();
  {
    v8f acc[16]; zero_acc<16>(acc);
    accumulate<16>(acc, Xb, XS, row0, 64, Wb, 72, lane);
    writeback<16>(acc, Hb, HS, row0, p.b1[0], lane, true);
  }

  // layers 1_1..1_4
  for (int i = 1; i < 5; ++i) {
    __syncthreads();
    stage_img(Wb, p.wimg + IMG_W1_1 + (i - 1) * (256 * 264), 256 * 264, tid);
#ifdef HAVE_ASYNC_LDS
    WAIT_ASYNC();
#endif
    __syncthreads();
    v8f acc[16]; zero_acc<16>(acc);
    accumulate<16>(acc, Hb, HS, row0, 256, Wb, WSTR, lane);
    writeback<16>(acc, Hb, HS, row0, p.b1[i], lane, true);
  }

  // layer 2_0 : [h|x_emb] -> 256
  __syncthreads();
  stage_img(Wb, p.wimg + IMG_W2_0M, 256 * 264, tid);
  stage_img(WXb, p.wimg + IMG_W2_0X, 256 * 72, tid);
#ifdef HAVE_ASYNC_LDS
  WAIT_ASYNC();
#endif
  __syncthreads();
  {
    v8f acc[16]; zero_acc<16>(acc);
    accumulate<16>(acc, Hb, HS, row0, 256, Wb, WSTR, lane);
    accumulate<16>(acc, Xb, XS, row0, 64, WXb, XS, lane);
    writeback<16>(acc, Hb, HS, row0, p.b2[0], lane, true);
  }

  // layers 2_1, 2_2
  for (int i = 1; i < 3; ++i) {
    __syncthreads();
    stage_img(Wb, p.wimg + IMG_W2_1 + (i - 1) * (256 * 264), 256 * 264, tid);
#ifdef HAVE_ASYNC_LDS
    WAIT_ASYNC();
#endif
    __syncthreads();
    v8f acc[16]; zero_acc<16>(acc);
    accumulate<16>(acc, Hb, HS, row0, 256, Wb, WSTR, lane);
    writeback<16>(acc, Hb, HS, row0, p.b2[i], lane, true);
  }

  // layer 2_3 : 256 -> 257 (zero rows prebaked into image)
  __syncthreads();
  stage_img(Wb, p.wimg + IMG_W2_3, 272 * 264, tid);
#ifdef HAVE_ASYNC_LDS
  WAIT_ASYNC();
#endif
  __syncthreads();

  sigma_and_head(p, Hb, Db, Wb, WXb, ray0, row0, lane, tid, true);
}

// ================= fallback kernel: f32 convert-on-stage =================
__global__ void __launch_bounds__(THREADS, 1)
nerf_fused_fallback(Params p) {
  extern __shared__ char smem[];
  _Float16* Hb  = (_Float16*)(smem + OFF_H);
  _Float16* Xb  = (_Float16*)(smem + OFF_X);
  _Float16* Db  = (_Float16*)(smem + OFF_D);
  _Float16* Wb  = (_Float16*)(smem + OFF_W);
  _Float16* WXb = (_Float16*)(smem + OFF_WX);

  const int tid  = threadIdx.x;
  const int lane = tid & 31;
  const int wave = tid >> 5;
  const int row0 = wave << 4;
  const int ray0 = blockIdx.x * RAYS_WG;

  posenc_body(p, Xb, Db, ray0, tid);

  __syncthreads();
  stage_w(Wb, WSTR, p.w1[0], 63, 0, 256, 256, 63, 6, tid);
  __syncthreads();
  {
    v8f acc[16]; zero_acc<16>(acc);
    accumulate<16>(acc, Xb, XS, row0, 64, Wb, WSTR, lane);
    writeback<16>(acc, Hb, HS, row0, p.b1[0], lane, true);
  }

  for (int i = 1; i < 5; ++i) {
    __syncthreads();
    stage_w4(Wb, WSTR, p.w1[i], 256, tid);
    __syncthreads();
    v8f acc[16]; zero_acc<16>(acc);
    accumulate<16>(acc, Hb, HS, row0, 256, Wb, WSTR, lane);
    writeback<16>(acc, Hb, HS, row0, p.b1[i], lane, true);
  }

  __syncthreads();
  stage_w(Wb, WSTR, p.w2[0], 319, 0, 256, 256, 256, 8, tid);
  stage_w(WXb, XS, p.w2[0], 319, 256, 256, 256, 63, 6, tid);
  __syncthreads();
  {
    v8f acc[16]; zero_acc<16>(acc);
    accumulate<16>(acc, Hb, HS, row0, 256, Wb, WSTR, lane);
    accumulate<16>(acc, Xb, XS, row0, 64, WXb, XS, lane);
    writeback<16>(acc, Hb, HS, row0, p.b2[0], lane, true);
  }

  for (int i = 1; i < 3; ++i) {
    __syncthreads();
    stage_w4(Wb, WSTR, p.w2[i], 256, tid);
    __syncthreads();
    v8f acc[16]; zero_acc<16>(acc);
    accumulate<16>(acc, Hb, HS, row0, 256, Wb, WSTR, lane);
    writeback<16>(acc, Hb, HS, row0, p.b2[i], lane, true);
  }

  __syncthreads();
  stage_w4(Wb, WSTR, p.w2[3], 257, tid);
  for (int idx = tid; idx < 15 * 256; idx += THREADS) {
    const int r = 257 + (idx >> 8), k = idx & 255;
    Wb[r * WSTR + k] = (_Float16)0.f;
  }
  __syncthreads();

  sigma_and_head(p, Hb, Db, Wb, WXb, ray0, row0, lane, tid, false);
}

// ================= prep kernel: f32 -> padded f16 images =================
__global__ void convert_w_kernel(_Float16* dst, const float* src, int srcStride,
                                 int srcOff, int srcRows, int K, int dstStride,
                                 int rows) {
  const int idx = blockIdx.x * blockDim.x + threadIdx.x;
  const int total = rows * dstStride;
  if (idx >= total) return;
  const int r = idx / dstStride, k = idx - r * dstStride;
  const float v = (r < srcRows && k < K) ? src[r * srcStride + srcOff + k] : 0.f;
  dst[idx] = (_Float16)v;
}

static inline void launch_convert(_Float16* dst, const float* src, int srcStride,
                                  int srcOff, int srcRows, int K, int dstStride,
                                  int rows, hipStream_t stream) {
  const int total = rows * dstStride;
  const int grid = (total + THREADS - 1) / THREADS;
  hipLaunchKernelGGL(convert_w_kernel, dim3(grid), dim3(THREADS), 0, stream,
                     dst, src, srcStride, srcOff, srcRows, K, dstStride, rows);
}

extern "C" void kernel_launch(void* const* d_in, const int* in_sizes, int n_in,
                              void* d_out, int out_size, void* d_ws, size_t ws_size,
                              hipStream_t stream) {
  (void)n_in; (void)out_size;
  Params p;
  p.xyz = (const float*)d_in[0];
  p.dir = (const float*)d_in[1];
  for (int i = 0; i < 5; ++i) {
    p.w1[i] = (const float*)d_in[2 + 2 * i];
    p.b1[i] = (const float*)d_in[3 + 2 * i];
  }
  for (int i = 0; i < 4; ++i) {
    p.w2[i] = (const float*)d_in[12 + 2 * i];
    p.b2[i] = (const float*)d_in[13 + 2 * i];
  }
  p.wr0 = (const float*)d_in[20];
  p.br0 = (const float*)d_in[21];
  p.wr1 = (const float*)d_in[22];
  p.br1 = (const float*)d_in[23];

  const int N = in_sizes[0] / 3;
  p.out_c = (float*)d_out;
  p.out_sigma = (float*)d_out + (size_t)3 * N;
  const int grid = N / RAYS_WG;

  if (ws_size >= WS_NEEDED) {
    _Float16* w = (_Float16*)d_ws;
    p.wimg = w;
    // f32 -> padded f16 LDS images (tiny; all hot in L2 afterwards)
    launch_convert(w + IMG_W1_0, p.w1[0], 63, 0, 256, 63, 72, 256, stream);
    for (int i = 1; i < 5; ++i)
      launch_convert(w + IMG_W1_1 + (i - 1) * (256 * 264), p.w1[i], 256, 0, 256, 256,
                     264, 256, stream);
    launch_convert(w + IMG_W2_0M, p.w2[0], 319, 0, 256, 256, 264, 256, stream);
    launch_convert(w + IMG_W2_0X, p.w2[0], 319, 256, 256, 63, 72, 256, stream);
    launch_convert(w + IMG_W2_1, p.w2[1], 256, 0, 256, 256, 264, 256, stream);
    launch_convert(w + IMG_W2_2, p.w2[2], 256, 0, 256, 256, 264, 256, stream);
    launch_convert(w + IMG_W2_3, p.w2[3], 256, 0, 257, 256, 264, 272, stream);
    launch_convert(w + IMG_WR0M, p.wr0, 283, 0, 128, 256, 264, 128, stream);
    launch_convert(w + IMG_WR0X, p.wr0, 283, 256, 128, 27, 40, 128, stream);

    hipLaunchKernelGGL(nerf_fused_f16w, dim3(grid), dim3(THREADS), LDS_BYTES, stream, p);
  } else {
    p.wimg = nullptr;
    hipLaunchKernelGGL(nerf_fused_fallback, dim3(grid), dim3(THREADS), LDS_BYTES, stream, p);
  }
}
